// MultiheadAttention_40261023433035
// MI455X (gfx1250) — matrix operand order repaired and checked by reading, not compiled, not run on hardware
//
#include <hip/hip_runtime.h>
#include <hip/hip_bf16.h>

// MI455X (gfx1250) multi-head causal attention, bf16 WMMA pipeline, v3.
// B=4, S=2048, D=1024, H=16, HD=64.
//
// Pipeline:
//  1) cvt_bf16        : query/key/value fp32 -> bf16 (one pass, vectorized)
//  2) wt_transpose    : Wq/Wk/Wv/Wo fp32 [K,N] -> bf16 [N,K] (LDS transpose)
//  3) gemm_bf16<0/1>  : Q = X*Wq -> [B,H,S,HD]; K likewise; V -> [B,H,HD,S]
//  4) attn            : flash attention, QK^T + online softmax + PV (WMMA)
//  5) gemm_bf16<2>    : out = Ao*Wo + bo -> fp32

typedef __attribute__((ext_vector_type(16))) __bf16         v16bf;
typedef __attribute__((ext_vector_type(8)))  float          v8f;
typedef __attribute__((ext_vector_type(4)))  float          float4v;
typedef __attribute__((ext_vector_type(8)))  unsigned short ushort8;

#define B_   4
#define S_   2048
#define D_   1024
#define H_   16
#define HD_  64
#define TOK  (B_ * S_)   // 8192 tokens

union FragBF { v16bf v; ushort8 h[2]; };

__device__ __forceinline__ unsigned short f2bfu(float f) {
  unsigned u = __float_as_uint(f);
  u += 0x7FFFu + ((u >> 16) & 1u);   // round-to-nearest-even
  return (unsigned short)(u >> 16);
}

// Build a 16-element bf16 fragment from two 16-byte chunks.
__device__ __forceinline__ v16bf ld_frag2(const unsigned short* p0,
                                          const unsigned short* p1) {
  FragBF f;
  f.h[0] = *(const ushort8*)p0;
  f.h[1] = *(const ushort8*)p1;
  return f.v;
}

__device__ __forceinline__ v8f wmma_bf16(v16bf a, v16bf b, v8f c) {
  return __builtin_amdgcn_wmma_f32_16x16x32_bf16(false, a, false, b,
                                                 (short)0, c, false, false);
}

// ---------------------------------------------------------------------------
// 1) fp32 -> bf16 flat conversion, 8 elements/thread, vectorized.
// ---------------------------------------------------------------------------
__global__ __launch_bounds__(256) void cvt_bf16_kernel(
    const float* __restrict__ X, unsigned short* __restrict__ Y)
{
  size_t base = ((size_t)blockIdx.x * 256 + threadIdx.x) * 8;
  float4v x0 = *(const float4v*)(X + base);
  float4v x1 = *(const float4v*)(X + base + 4);
  ushort8 y;
  #pragma unroll
  for (int i = 0; i < 4; ++i) y[i]     = f2bfu(x0[i]);
  #pragma unroll
  for (int i = 0; i < 4; ++i) y[i + 4] = f2bfu(x1[i]);
  *(ushort8*)(Y + base) = y;
}

// ---------------------------------------------------------------------------
// 2) Weight convert+transpose: W fp32 [K=1024][N=1024] -> Wt bf16 [N][K].
// ---------------------------------------------------------------------------
__global__ __launch_bounds__(256) void wt_transpose_kernel(
    const float* __restrict__ W, unsigned short* __restrict__ Wt)
{
  __shared__ __align__(16) unsigned short t[64][72];
  const int tid = threadIdx.x;
  const int k0 = blockIdx.x * 64, n0 = blockIdx.y * 64;
  #pragma unroll
  for (int i = 0; i < 16; ++i) {        // read coalesced along n
    int id = i * 256 + tid;
    int r = id >> 6, c = id & 63;       // r = k-local, c = n-local
    t[c][r] = f2bfu(W[(size_t)(k0 + r) * D_ + (n0 + c)]);
  }
  __syncthreads();
  #pragma unroll
  for (int i = 0; i < 16; ++i) {        // write coalesced along k
    int id = i * 256 + tid;
    int r = id >> 6, c = id & 63;       // r = n-local, c = k-local
    Wt[(size_t)(n0 + r) * D_ + (k0 + c)] = t[r][c];
  }
}

// ---------------------------------------------------------------------------
// 3/5) GEMM: C[64x128 tile] = A[bf16, TOKxD] * Wt^T (Wt bf16 [N][K]).
// MODE 0: bf16 out, [B,H,S,HD] layout (Q, K)
// MODE 1: bf16 out, [B,H,HD,S] layout (V transposed)
// MODE 2: fp32 out + bias, row-major [TOK, D] (output projection)
// 128 threads (4 waves); each wave: 16 rows x 128 cols, 8 WMMA per K-step.
// ---------------------------------------------------------------------------
template <int MODE>
__global__ __launch_bounds__(128) void gemm_bf16_kernel(
    const unsigned short* __restrict__ A, const unsigned short* __restrict__ Wt,
    const float* __restrict__ bias, unsigned short* __restrict__ outb,
    float* __restrict__ outf)
{
  __shared__ __align__(16) unsigned short As[64][40];   // [m][k], 80B rows
  __shared__ __align__(16) unsigned short Bs[128][40];  // [n][k]
  const int tid  = threadIdx.x;
  const int wave = tid >> 5, lane = tid & 31;
  const int lr   = lane & 15, hi = lane >> 4;
  const int m0 = blockIdx.y * 64, n0 = blockIdx.x * 128;

  v8f acc[8];
  #pragma unroll
  for (int nt = 0; nt < 8; ++nt)
    acc[nt] = (v8f){0.f,0.f,0.f,0.f,0.f,0.f,0.f,0.f};

  const int kA = hi ? 8 : 0;
  const int kB = hi ? 8 : 0;

  for (int k0 = 0; k0 < D_; k0 += 32) {
    #pragma unroll
    for (int i = 0; i < 2; ++i) {                 // 64x32 A tile, b128 copies
      int c = i * 128 + tid;
      int r = c >> 2, ch = (c & 3) * 8;
      *(ushort8*)&As[r][ch] = *(const ushort8*)(A + (size_t)(m0 + r) * D_ + k0 + ch);
    }
    #pragma unroll
    for (int i = 0; i < 4; ++i) {                 // 128x32 B tile
      int c = i * 128 + tid;
      int r = c >> 2, ch = (c & 3) * 8;
      *(ushort8*)&Bs[r][ch] = *(const ushort8*)(Wt + (size_t)(n0 + r) * D_ + k0 + ch);
    }
    __syncthreads();
    const int arow = wave * 16 + lr;
    v16bf a = ld_frag2(&As[arow][kA], &As[arow][kA + 16]);
    #pragma unroll
    for (int g = 0; g < 2; ++g) {                 // two groups of 4 WMMAs
      v16bf bfr[4];
      #pragma unroll
      for (int j = 0; j < 4; ++j) {
        int nt = g * 4 + j;
        bfr[j] = ld_frag2(&Bs[nt * 16 + lr][kB], &Bs[nt * 16 + lr][kB + 16]);
      }
      __builtin_amdgcn_sched_barrier(0);          // loads first, then WMMAs
      #pragma unroll
      for (int j = 0; j < 4; ++j)
        acc[g * 4 + j] = wmma_bf16(a, bfr[j], acc[g * 4 + j]);
      __builtin_amdgcn_sched_barrier(0);
    }
    __syncthreads();
  }

  #pragma unroll
  for (int nt = 0; nt < 8; ++nt) {
    #pragma unroll
    for (int r = 0; r < 8; ++r) {
      int gm = m0 + wave * 16 + r + hi * 8;    // token
      int gn = n0 + nt * 16 + lr;              // head*HD + d  (or plain col)
      if (MODE == 2) {
        outf[(size_t)gm * D_ + gn] = acc[nt][r] + bias[gn];
      } else {
        int bb = gm >> 11, ss = gm & (S_ - 1);
        int hh = gn >> 6,  dd = gn & (HD_ - 1);
        size_t idx = (MODE == 0)
          ? ((size_t)(bb * H_ + hh) * S_  + ss) * HD_ + dd
          : ((size_t)(bb * H_ + hh) * HD_ + dd) * S_  + ss;
        outb[idx] = f2bfu(acc[nt][r]);
      }
    }
  }
}

// ---------------------------------------------------------------------------
// 4) Flash attention: 1 WG (4 waves) = 64 queries of one (b,h).
// Causal: only blockIdx.x+1 key blocks of 64 are touched.
// ---------------------------------------------------------------------------
__global__ __launch_bounds__(128) void attn_kernel(
    const unsigned short* __restrict__ Qw,   // [B,H,S,HD]
    const unsigned short* __restrict__ Kw,   // [B,H,S,HD]
    const unsigned short* __restrict__ Vw,   // [B,H,HD,S]
    unsigned short* __restrict__ Ow)         // [B,S,H*HD]
{
  __shared__ __align__(16) unsigned short Ks[64][72];     // [key][d]
  __shared__ __align__(16) unsigned short Vs[64][72];     // [d][key]
  __shared__ __align__(16) unsigned short Ps[4][16][72];  // per-wave P tile
  const int tid  = threadIdx.x;
  const int wave = tid >> 5, lane = tid & 31;
  const int lr   = lane & 15, hi = lane >> 4;
  const int qblk = blockIdx.x, bh = blockIdx.y;

  const unsigned short* Qp = Qw + (size_t)bh * S_ * HD_;
  const unsigned short* Kp = Kw + (size_t)bh * S_ * HD_;
  const unsigned short* Vp = Vw + (size_t)bh * HD_ * S_;

  const int kA = hi ? 8 : 0;
  const int kB = hi ? 8 : 0;

  // Q A-fragments (k-dim = d, two 32-wide steps), loaded once.
  const int qrow = qblk * 64 + wave * 16 + lr;
  v16bf qfrag[2];
  #pragma unroll
  for (int ks = 0; ks < 2; ++ks)
    qfrag[ks] = ld_frag2(Qp + (size_t)qrow * HD_ + ks * 32 + kA,
                         Qp + (size_t)qrow * HD_ + ks * 32 + kA + 16);

  v8f o[4];
  #pragma unroll
  for (int dt = 0; dt < 4; ++dt)
    o[dt] = (v8f){0.f,0.f,0.f,0.f,0.f,0.f,0.f,0.f};
  float m[8], l[8];
  #pragma unroll
  for (int r = 0; r < 8; ++r) { m[r] = -1e30f; l[r] = 0.f; }

  for (int kb = 0; kb <= qblk; ++kb) {
    __syncthreads();
    #pragma unroll
    for (int i = 0; i < 4; ++i) {           // 64x64 bf16 K and V tiles
      int c = i * 128 + tid;
      int r = c >> 3, ch = (c & 7) * 8;
      *(ushort8*)&Ks[r][ch] = *(const ushort8*)(Kp + (size_t)(kb * 64 + r) * HD_ + ch);
      *(ushort8*)&Vs[r][ch] = *(const ushort8*)(Vp + (size_t)r * S_ + kb * 64 + ch);
    }
    if (kb < qblk) {                        // prefetch next key block
      int r = tid >> 1, half = (tid & 1) * 32;
      __builtin_prefetch(Kp + (size_t)((kb + 1) * 64 + r) * HD_ + half, 0, 3);
      __builtin_prefetch(Vp + (size_t)r * S_ + (kb + 1) * 64 + half, 0, 3);
    }
    __syncthreads();

    // --- scores = Q K^T / 8, causal mask ---
    float st[4][8];
    #pragma unroll
    for (int nt = 0; nt < 4; ++nt) {
      v8f sc = (v8f){0.f,0.f,0.f,0.f,0.f,0.f,0.f,0.f};
      const unsigned short* bp = &Ks[nt * 16 + lr][0];
      v16bf bk[2];
      #pragma unroll
      for (int ds = 0; ds < 2; ++ds)
        bk[ds] = ld_frag2(bp + ds * 32 + kB, bp + ds * 32 + kB + 16);
      __builtin_amdgcn_sched_barrier(0);
      #pragma unroll
      for (int ds = 0; ds < 2; ++ds)
        sc = wmma_bf16(qfrag[ds], bk[ds], sc);
      __builtin_amdgcn_sched_barrier(0);
      int key = kb * 64 + nt * 16 + lr;
      #pragma unroll
      for (int r = 0; r < 8; ++r) {
        int q = qblk * 64 + wave * 16 + r + hi * 8;
        st[nt][r] = (key > q) ? -1e30f : sc[r] * 0.125f;
      }
    }

    // --- online softmax update (row reductions within 16-lane half groups) ---
    float fac[8];
    #pragma unroll
    for (int r = 0; r < 8; ++r) {
      float vmax = fmaxf(fmaxf(st[0][r], st[1][r]), fmaxf(st[2][r], st[3][r]));
      vmax = fmaxf(vmax, __shfl_xor(vmax, 1, 32));
      vmax = fmaxf(vmax, __shfl_xor(vmax, 2, 32));
      vmax = fmaxf(vmax, __shfl_xor(vmax, 4, 32));
      vmax = fmaxf(vmax, __shfl_xor(vmax, 8, 32));
      float mnew = fmaxf(m[r], vmax);
      fac[r] = __expf(m[r] - mnew);
      float psum = 0.f;
      int prow = r + hi * 8;
      #pragma unroll
      for (int nt = 0; nt < 4; ++nt) {
        float p = __expf(st[nt][r] - mnew);
        Ps[wave][prow][nt * 16 + lr] = f2bfu(p);
        psum += p;
      }
      psum += __shfl_xor(psum, 1, 32);
      psum += __shfl_xor(psum, 2, 32);
      psum += __shfl_xor(psum, 4, 32);
      psum += __shfl_xor(psum, 8, 32);
      l[r] = l[r] * fac[r] + psum;
      m[r] = mnew;
    }
    #pragma unroll
    for (int dt = 0; dt < 4; ++dt)
      #pragma unroll
      for (int r = 0; r < 8; ++r) o[dt][r] *= fac[r];

    __syncthreads();   // make cross-lane P stores visible to fragment loads

    // --- O += P V  (k-dim = key) ---
    v16bf pa[2];
    #pragma unroll
    for (int ks = 0; ks < 2; ++ks)
      pa[ks] = ld_frag2(&Ps[wave][lr][ks * 32 + kA],
                        &Ps[wave][lr][ks * 32 + kA + 16]);
    #pragma unroll
    for (int dt = 0; dt < 4; ++dt) {
      const unsigned short* vp = &Vs[dt * 16 + lr][0];
      v16bf vb[2];
      #pragma unroll
      for (int ks = 0; ks < 2; ++ks)
        vb[ks] = ld_frag2(vp + ks * 32 + kB, vp + ks * 32 + kB + 16);
      __builtin_amdgcn_sched_barrier(0);
      #pragma unroll
      for (int ks = 0; ks < 2; ++ks)
        o[dt] = wmma_bf16(pa[ks], vb[ks], o[dt]);
      __builtin_amdgcn_sched_barrier(0);
    }
  }

  // --- finalize: divide by l, scatter to [B,S,H*HD] bf16 ---
  const int bb = bh >> 4, hh = bh & 15;
  #pragma unroll
  for (int r = 0; r < 8; ++r) {
    float inv = 1.0f / l[r];
    int q = qblk * 64 + wave * 16 + r + hi * 8;
    size_t tok = (size_t)bb * S_ + q;
    #pragma unroll
    for (int dt = 0; dt < 4; ++dt)
      Ow[tok * D_ + hh * HD_ + dt * 16 + lr] = f2bfu(o[dt][r] * inv);
  }
}

// ---------------------------------------------------------------------------
extern "C" void kernel_launch(void* const* d_in, const int* in_sizes, int n_in,
                              void* d_out, int out_size, void* d_ws, size_t ws_size,
                              hipStream_t stream) {
  (void)in_sizes; (void)n_in; (void)out_size; (void)ws_size;
  const float* query = (const float*)d_in[0];
  const float* key_i = (const float*)d_in[1];
  const float* value = (const float*)d_in[2];
  // d_in[3] = boolean causal mask: applied analytically in attn_kernel.
  const float* Wq = (const float*)d_in[4];
  const float* Wk = (const float*)d_in[5];
  const float* Wv = (const float*)d_in[6];
  const float* Wo = (const float*)d_in[7];
  const float* bo = (const float*)d_in[8];
  float* out = (float*)d_out;

  // Workspace layout (bf16 = unsigned short):
  //   Qb/Kb/Vb/Ao : 4 x TOK*D  (64 MB)
  //   Xq/Xk/Xv    : 3 x TOK*D  (48 MB)
  //   Wqt/Wkt/Wvt/Wot : 4 x D*D (8 MB)
  unsigned short* p  = (unsigned short*)d_ws;
  unsigned short* Qb  = p;  p += (size_t)TOK * D_;
  unsigned short* Kb  = p;  p += (size_t)TOK * D_;
  unsigned short* Vb  = p;  p += (size_t)TOK * D_;
  unsigned short* Ao  = p;  p += (size_t)TOK * D_;
  unsigned short* Xq  = p;  p += (size_t)TOK * D_;
  unsigned short* Xk  = p;  p += (size_t)TOK * D_;
  unsigned short* Xv  = p;  p += (size_t)TOK * D_;
  unsigned short* Wqt = p;  p += (size_t)D_ * D_;
  unsigned short* Wkt = p;  p += (size_t)D_ * D_;
  unsigned short* Wvt = p;  p += (size_t)D_ * D_;
  unsigned short* Wot = p;

  const int cvtBlocks = (int)(((size_t)TOK * D_) / (256 * 8));   // 4096
  cvt_bf16_kernel<<<cvtBlocks, 256, 0, stream>>>(query, Xq);
  cvt_bf16_kernel<<<cvtBlocks, 256, 0, stream>>>(key_i, Xk);
  cvt_bf16_kernel<<<cvtBlocks, 256, 0, stream>>>(value, Xv);

  dim3 tGrid(D_ / 64, D_ / 64);   // (16,16)
  wt_transpose_kernel<<<tGrid, 256, 0, stream>>>(Wq, Wqt);
  wt_transpose_kernel<<<tGrid, 256, 0, stream>>>(Wk, Wkt);
  wt_transpose_kernel<<<tGrid, 256, 0, stream>>>(Wv, Wvt);
  wt_transpose_kernel<<<tGrid, 256, 0, stream>>>(Wo, Wot);

  dim3 gemmGrid(D_ / 128, TOK / 64);   // (8, 128)
  gemm_bf16_kernel<0><<<gemmGrid, 128, 0, stream>>>(Xq, Wqt, nullptr, Qb, nullptr);
  gemm_bf16_kernel<0><<<gemmGrid, 128, 0, stream>>>(Xk, Wkt, nullptr, Kb, nullptr);
  gemm_bf16_kernel<1><<<gemmGrid, 128, 0, stream>>>(Xv, Wvt, nullptr, Vb, nullptr);

  attn_kernel<<<dim3(S_ / 64, B_ * H_), 128, 0, stream>>>(Qb, Kb, Vb, Ao);

  gemm_bf16_kernel<2><<<gemmGrid, 128, 0, stream>>>(Ao, Wot, bo, nullptr, out);
}
